// MultiHeadAttention_44212393345709
// MI455X (gfx1250) — matrix operand order repaired and checked by reading, not compiled, not run on hardware
//
#include <hip/hip_runtime.h>

// ---------------------------------------------------------------------------
// MI455X (gfx1250, wave32) fused pre-LN causal MHA block.
// GEMMs: V_WMMA_F32_16X16X32_F16, 32x64 register tile per wave, B staged in
// LDS per 4-wave workgroup, async-to-LDS double buffering (ASYNCcnt).
// ---------------------------------------------------------------------------

typedef __attribute__((ext_vector_type(16))) _Float16 v16h;
typedef __attribute__((ext_vector_type(8)))  _Float16 v8h;
typedef __attribute__((ext_vector_type(8)))  float    v8f;

#define DM     1024      // d_model
#define T_SEQ  2048
#define NB     2
#define NH     16
#define DH     64
#define BT     (NB * T_SEQ)   // 4096 rows

#if __has_builtin(__builtin_amdgcn_global_load_async_to_lds_b128) && \
    __has_builtin(__builtin_amdgcn_s_wait_asynccnt)
#define HAVE_ASYNC_LDS 1
// Builtin signature (verified by hipcc diagnostic): (global int4*, lds int4*,
// imm offset, imm cpol).
typedef int v4i_ __attribute__((vector_size(4 * sizeof(int))));
typedef __attribute__((address_space(1))) v4i_ gv4i;
typedef __attribute__((address_space(3))) v4i_ lv4i;
#else
#define HAVE_ASYNC_LDS 0
#endif

static __device__ __forceinline__ v8f wmma16(v16h a, v16h b, v8f c) {
  // D = A(16x32 f16) * B(32x16 f16) + C(16x16 f32)
  return __builtin_amdgcn_wmma_f32_16x16x32_f16(
      /*neg_a=*/false, a, /*neg_b=*/false, b,
      /*c_mod=*/(short)0, c, /*reuse_a=*/false, /*reuse_b=*/false);
}

static __device__ __forceinline__ v16h cat8(v8h lo, v8h hi) {
  return __builtin_shufflevector(lo, hi, 0, 1, 2, 3, 4, 5, 6, 7,
                                         8, 9, 10, 11, 12, 13, 14, 15);
}

// A-fragment (16x32, 16-bit): lane holds row (lane&15);
// elements 0..15 -> K = hl*16 + i.
static __device__ __forceinline__ v16h load_a_frag(const _Float16* p, int hl) {
  v8h lo = *(const v8h*)(p + hl * 16);
  v8h hi = *(const v8h*)(p + hl * 16 + 8);
  return cat8(lo, hi);
}

// B-fragment (32x16, 16-bit): lane holds col (lane&15);
// elements 0..15 -> K = hl*16 + i  (p must already include the hl*16 offset).
static __device__ __forceinline__ v16h load_b_frag(const _Float16* p) {
  v8h lo = *(const v8h*)(p);
  v8h hi = *(const v8h*)(p + 8);
  return cat8(lo, hi);
}

// ---------------------------------------------------------------------------
// 1) pre-LayerNorm: xn32 (residual) + xn16 (GEMM input). One block per row.
// ---------------------------------------------------------------------------
__global__ void ln_kernel(const float* __restrict__ x,
                          const float* __restrict__ gamma,
                          const float* __restrict__ beta,
                          float* __restrict__ xn32,
                          _Float16* __restrict__ xn16) {
  const int row = blockIdx.x;         // 0..BT-1
  const int tid = threadIdx.x;        // 0..255 (8 waves)
  const float* xr = x + (size_t)row * DM;

  float4 v = ((const float4*)xr)[tid];          // 256 * 4 = 1024
  float s  = v.x + v.y + v.z + v.w;
  float ss = v.x * v.x + v.y * v.y + v.z * v.z + v.w * v.w;
  for (int m = 16; m >= 1; m >>= 1) {
    s  += __shfl_xor(s,  m, 32);
    ss += __shfl_xor(ss, m, 32);
  }
  __shared__ float red0[8];
  __shared__ float red1[8];
  const int wid = tid >> 5;
  if ((tid & 31) == 0) { red0[wid] = s; red1[wid] = ss; }
  __syncthreads();
  float st = 0.f, sst = 0.f;
  for (int i = 0; i < 8; ++i) { st += red0[i]; sst += red1[i]; }

  const float mu   = st * (1.0f / DM);
  const float var  = sst * (1.0f / DM) - mu * mu;
  const float rstd = rsqrtf(var + 1e-5f);

  const int base = tid * 4;
  float* o32 = xn32 + (size_t)row * DM + base;
  _Float16* o16 = xn16 + (size_t)row * DM + base;
  const float* xv = &v.x;
  for (int c = 0; c < 4; ++c) {
    const float y = (xv[c] - mu) * rstd * gamma[base + c] + beta[base + c];
    o32[c] = y;
    o16[c] = (_Float16)y;
  }
}

// ---------------------------------------------------------------------------
// 2) Weight transpose + f32->f16:  Wt[n][k] = W[k][n]  (N-major for B frags)
// ---------------------------------------------------------------------------
__global__ void wtrans_kernel(const float* __restrict__ W,
                              _Float16* __restrict__ Wt) {
  const int n = blockIdx.x * 16 + threadIdx.x;
  const int k = blockIdx.y * 16 + threadIdx.y;
  Wt[(size_t)n * DM + k] = (_Float16)W[(size_t)k * DM + n];
}

// ---------------------------------------------------------------------------
// 3) Tiled GEMM: block = 4 waves = 128 threads, tile 128M x 64N.
//    Each wave: 32M x 64N (8 accumulators).  B tile (64N x 32K f16 = 4KB)
//    staged in LDS, shared by all waves; async double-buffered when possible.
//    MODE 0: out16 row-major [BT][DM]           (Q, K)
//    MODE 1: out16 = Vt[b][DM][T_SEQ]           (V, time-minor)
//    MODE 2: out32 = acc + bo[col] + xn32       (output projection, fp32)
// ---------------------------------------------------------------------------
template <int MODE>
__global__ void gemm_tile_kernel(const _Float16* __restrict__ A,
                                 const _Float16* __restrict__ Bt,
                                 _Float16* __restrict__ out16,
                                 const float* __restrict__ bo,
                                 const float* __restrict__ xn32,
                                 float* __restrict__ out32) {
  const int tid  = threadIdx.x;         // 0..127
  const int wave = tid >> 5;            // 0..3
  const int lane = tid & 31;
  const int r    = lane & 15;
  const int hl   = lane >> 4;
  const int m0   = blockIdx.x * 128 + wave * 32;
  const int n0   = blockIdx.y * 64;

  __shared__ __align__(16) _Float16 bbuf[2][64 * 32];   // 2 x 4KB

  v8f acc0[4] = {};   // rows m0 +  0..15
  v8f acc1[4] = {};   // rows m0 + 16..31

  const _Float16* arow0 = A + (size_t)(m0 + r) * DM;
  const _Float16* arow1 = A + (size_t)(m0 + 16 + r) * DM;

  // Per-thread staging source: row n = tid>>1 of the B tile, half = tid&1.
  const int srow = tid >> 1;
  const int shalf = (tid & 1) * 16;
  const _Float16* bsrc = Bt + (size_t)(n0 + srow) * DM + shalf;
  const int ldst = srow * 32 + shalf;   // LDS offset in halves

  auto compute = [&](int k0, const _Float16* bb) {
    const v16h a0 = load_a_frag(arow0 + k0, hl);
    const v16h a1 = load_a_frag(arow1 + k0, hl);
#pragma unroll
    for (int nt = 0; nt < 4; ++nt) {
      const v16h bf = load_b_frag(bb + (nt * 16 + r) * 32 + hl * 16);
      acc0[nt] = wmma16(a0, bf, acc0[nt]);
      acc1[nt] = wmma16(a1, bf, acc1[nt]);
    }
  };

#if HAVE_ASYNC_LDS
  // Async DMA staging (ASYNCcnt-tracked), 2-deep pipeline.
  auto stage = [&](int k0, _Float16* dst) {
    __builtin_amdgcn_global_load_async_to_lds_b128(
        (gv4i*)(bsrc + k0), (lv4i*)(dst + ldst), 0, 0);
    __builtin_amdgcn_global_load_async_to_lds_b128(
        (gv4i*)(bsrc + k0 + 8), (lv4i*)(dst + ldst + 8), 0, 0);
  };
  stage(0, bbuf[0]);
  int p = 0;
  for (int k0 = 0; k0 < DM - 32; k0 += 32) {
    stage(k0 + 32, bbuf[p ^ 1]);            // prefetch next tile (async)
    __builtin_amdgcn_s_wait_asynccnt(2);    // wait for *current* tile only
    __syncthreads();
    compute(k0, bbuf[p]);
    __syncthreads();                        // all readers done before overwrite
    p ^= 1;
  }
  __builtin_amdgcn_s_wait_asynccnt(0);
  __syncthreads();
  compute(DM - 32, bbuf[p]);
#else
  // Synchronous staging fallback.
  for (int k0 = 0; k0 < DM; k0 += 32) {
    __syncthreads();
    *(v8h*)(&bbuf[0][ldst])     = *(const v8h*)(bsrc + k0);
    *(v8h*)(&bbuf[0][ldst + 8]) = *(const v8h*)(bsrc + k0 + 8);
    __syncthreads();
    compute(k0, bbuf[0]);
  }
#endif

  // ---- epilogue (C layout: row m = j + 8*hl, col n = r) --------------------
  // No runtime-selected pointers into the accumulator arrays (keeps SROA
  // intact so accumulators never touch scratch).
#pragma unroll
  for (int nt = 0; nt < 4; ++nt) {
    const int col = n0 + nt * 16 + r;
    if (MODE == 0) {
      _Float16* o0 = out16 + (size_t)(m0 + 8 * hl) * DM + col;
      _Float16* o1 = out16 + (size_t)(m0 + 16 + 8 * hl) * DM + col;
#pragma unroll
      for (int j = 0; j < 8; ++j) {
        o0[(size_t)j * DM] = (_Float16)acc0[nt][j];
        o1[(size_t)j * DM] = (_Float16)acc1[nt][j];
      }
    } else if (MODE == 1) {
      // Vt[b][col][t], b = row / T_SEQ, t = row % T_SEQ. All 32 rows of this
      // wave tile sit in the same batch (m0 is a multiple of 32).
      const int bidx = m0 >> 11;
      const int t0   = m0 & (T_SEQ - 1);
      _Float16* ocol =
          out16 + (size_t)bidx * DM * T_SEQ + (size_t)col * T_SEQ + t0;
#pragma unroll
      for (int j = 0; j < 8; ++j) {
        ocol[8 * hl + j]      = (_Float16)acc0[nt][j];
        ocol[16 + 8 * hl + j] = (_Float16)acc1[nt][j];
      }
    } else {
      const float bias = bo[col];
      const float* x0 = xn32 + (size_t)(m0 + 8 * hl) * DM + col;
      const float* x1 = xn32 + (size_t)(m0 + 16 + 8 * hl) * DM + col;
      float* o0 = out32 + (size_t)(m0 + 8 * hl) * DM + col;
      float* o1 = out32 + (size_t)(m0 + 16 + 8 * hl) * DM + col;
#pragma unroll
      for (int j = 0; j < 8; ++j) {
        o0[(size_t)j * DM] = acc0[nt][j] + bias + x0[(size_t)j * DM];
        o1[(size_t)j * DM] = acc1[nt][j] + bias + x1[(size_t)j * DM];
      }
    }
  }
}

// ---------------------------------------------------------------------------
// 4) Causal flash attention: one wave per (qtile, head, batch).
//    32 keys / iteration, fp32 online softmax, O accumulated via WMMA.
// ---------------------------------------------------------------------------
__global__ void attn_kernel(const _Float16* __restrict__ Q,   // [BT][DM] rm
                            const _Float16* __restrict__ K,   // [BT][DM] rm
                            const _Float16* __restrict__ Vt,  // [NB][DM][T]
                            _Float16* __restrict__ O) {       // [BT][DM] rm
  const int qt   = blockIdx.x;          // 0..T/16-1
  const int h    = blockIdx.y;          // 0..NH-1
  const int b    = blockIdx.z;          // 0..NB-1
  const int lane = threadIdx.x & 31;
  const int r    = lane & 15;
  const int hl   = lane >> 4;
  const int q0   = qt * 16;
  const int hoff = h * DH;

  const _Float16* Qb = Q  + (size_t)(b * T_SEQ + q0) * DM + hoff;
  const _Float16* Kb = K  + (size_t)(b * T_SEQ) * DM + hoff;
  const _Float16* Vb = Vt + (size_t)b * DM * T_SEQ + (size_t)hoff * T_SEQ;

  __shared__ __align__(32) _Float16 pbuf[16 * 32];   // P tile (row-major, ld=32)

  // Q fragments, constant over the key loop: K-chunks kc = 0 and 32 of D_HEAD.
  v16h qf[2];
#pragma unroll
  for (int kc = 0; kc < 2; ++kc)
    qf[kc] = load_a_frag(Qb + (size_t)r * DM + kc * 32, hl);

  v8f o_acc[4] = {};
  float mrow[8], lrow[8];
#pragma unroll
  for (int j = 0; j < 8; ++j) { mrow[j] = -3.0e38f; lrow[j] = 0.f; }

  const int qmax = q0 + 15;
  for (int kb = 0; kb <= qmax; kb += 32) {
    // ---- S = (Q K^T) * scale, two 16-key subtiles --------------------------
    v8f s[2];
#pragma unroll
    for (int st = 0; st < 2; ++st) {
      v8f acc = {};
      const int key_base = kb + st * 16;
#pragma unroll
      for (int kc = 0; kc < 2; ++kc) {
        // B = K^T chunk: col c = key_base + r ; K-dim = kc*32 + hl*16 + i
        const _Float16* pk =
            Kb + (size_t)(key_base + r) * DM + kc * 32 + hl * 16;
        acc = wmma16(qf[kc], load_b_frag(pk), acc);
      }
      // scale + causal mask (C layout: row m = j + 8*hl, col n = r)
#pragma unroll
      for (int j = 0; j < 8; ++j) {
        if (key_base + r > q0 + j + 8 * hl) acc[j] = -3.0e38f;
        else                                acc[j] *= 0.125f;  // 1/sqrt(64)
      }
      s[st] = acc;
    }

    // ---- online softmax over the 32 new keys -------------------------------
    float alpha[8];
#pragma unroll
    for (int j = 0; j < 8; ++j) {
      float tmax = fmaxf(s[0][j], s[1][j]);
      for (int m = 8; m >= 1; m >>= 1) tmax = fmaxf(tmax, __shfl_xor(tmax, m, 32));
      const float mnew = fmaxf(mrow[j], tmax);
      alpha[j] = __expf(mrow[j] - mnew);
      mrow[j]  = mnew;
    }
#pragma unroll
    for (int j = 0; j < 8; ++j) {
      const float p0 = __expf(s[0][j] - mrow[j]);
      const float p1 = __expf(s[1][j] - mrow[j]);
      s[0][j] = p0;
      s[1][j] = p1;
      float rs = p0 + p1;
      for (int m = 8; m >= 1; m >>= 1) rs += __shfl_xor(rs, m, 32);
      lrow[j] = lrow[j] * alpha[j] + rs;
#pragma unroll
      for (int nt = 0; nt < 4; ++nt) o_acc[nt][j] *= alpha[j];
    }

    // ---- P (C layout) -> LDS -> A-fragment layout --------------------------
    __syncthreads();
#pragma unroll
    for (int st = 0; st < 2; ++st)
#pragma unroll
      for (int j = 0; j < 8; ++j)
        pbuf[(j + 8 * hl) * 32 + st * 16 + r] = (_Float16)s[st][j];
    __syncthreads();

    const v16h pf = load_a_frag(&pbuf[r * 32], hl);

    // ---- O += P * V --------------------------------------------------------
#pragma unroll
    for (int nt = 0; nt < 4; ++nt) {
      const _Float16* pv =
          Vb + (size_t)(nt * 16 + r) * T_SEQ + kb + hl * 16;
      o_acc[nt] = wmma16(pf, load_b_frag(pv), o_acc[nt]);
    }
  }

  // ---- normalize and store ------------------------------------------------
  float invl[8];
#pragma unroll
  for (int j = 0; j < 8; ++j) invl[j] = 1.0f / lrow[j];
#pragma unroll
  for (int nt = 0; nt < 4; ++nt) {
    _Float16* orow =
        O + (size_t)(b * T_SEQ + q0 + 8 * hl) * DM + hoff + nt * 16 + r;
#pragma unroll
    for (int j = 0; j < 8; ++j)
      orow[(size_t)j * DM] = (_Float16)(o_acc[nt][j] * invl[j]);
  }
}

// ---------------------------------------------------------------------------
// Launch
// ---------------------------------------------------------------------------
extern "C" void kernel_launch(void* const* d_in, const int* in_sizes, int n_in,
                              void* d_out, int out_size, void* d_ws, size_t ws_size,
                              hipStream_t stream) {
  const float* x     = (const float*)d_in[0];
  const float* Wq    = (const float*)d_in[1];
  const float* Wk    = (const float*)d_in[2];
  const float* Wv    = (const float*)d_in[3];
  const float* Wo    = (const float*)d_in[4];
  const float* bo    = (const float*)d_in[5];
  const float* gamma = (const float*)d_in[6];
  const float* beta  = (const float*)d_in[7];
  float* out = (float*)d_out;

  char* ws = (char*)d_ws;
  size_t off = 0;
  auto alloc = [&](size_t bytes) -> void* {
    void* p = ws + off;
    off += (bytes + 255) & ~(size_t)255;
    return p;
  };

  float*    xn32 = (float*)   alloc((size_t)BT * DM * sizeof(float));
  _Float16* xn16 = (_Float16*)alloc((size_t)BT * DM * sizeof(_Float16));
  _Float16* Wqt  = (_Float16*)alloc((size_t)DM * DM * sizeof(_Float16));
  _Float16* Wkt  = (_Float16*)alloc((size_t)DM * DM * sizeof(_Float16));
  _Float16* Wvt  = (_Float16*)alloc((size_t)DM * DM * sizeof(_Float16));
  _Float16* Wot  = (_Float16*)alloc((size_t)DM * DM * sizeof(_Float16));
  _Float16* Qh   = (_Float16*)alloc((size_t)BT * DM * sizeof(_Float16));
  _Float16* Kh   = (_Float16*)alloc((size_t)BT * DM * sizeof(_Float16));
  _Float16* Vt   = (_Float16*)alloc((size_t)NB * DM * T_SEQ * sizeof(_Float16));
  _Float16* Oh   = (_Float16*)alloc((size_t)BT * DM * sizeof(_Float16));
  (void)ws_size;

  // 1) LayerNorm
  ln_kernel<<<BT, 256, 0, stream>>>(x, gamma, beta, xn32, xn16);

  // 2) Weight transposes (f32 -> f16, N-major)
  {
    dim3 g(DM / 16, DM / 16), blk(16, 16);
    wtrans_kernel<<<g, blk, 0, stream>>>(Wq, Wqt);
    wtrans_kernel<<<g, blk, 0, stream>>>(Wk, Wkt);
    wtrans_kernel<<<g, blk, 0, stream>>>(Wv, Wvt);
    wtrans_kernel<<<g, blk, 0, stream>>>(Wo, Wot);
  }

  // 3) QKV projections (V written time-minor transposed)
  {
    dim3 g(BT / 128, DM / 64);
    gemm_tile_kernel<0><<<g, 128, 0, stream>>>(xn16, Wqt, Qh, nullptr, nullptr,
                                               nullptr);
    gemm_tile_kernel<0><<<g, 128, 0, stream>>>(xn16, Wkt, Kh, nullptr, nullptr,
                                               nullptr);
    gemm_tile_kernel<1><<<g, 128, 0, stream>>>(xn16, Wvt, Vt, nullptr, nullptr,
                                               nullptr);
  }

  // 4) Causal attention
  {
    dim3 g(T_SEQ / 16, NH, NB);
    attn_kernel<<<g, 32, 0, stream>>>(Qh, Kh, Vt, Oh);
  }

  // 5) Output projection + bias + residual
  {
    dim3 g(BT / 128, DM / 64);
    gemm_tile_kernel<2><<<g, 128, 0, stream>>>(Oh, Wot, nullptr, bo, xn32, out);
  }
}
